// MeshGraphNet_78168404787528
// MI455X (gfx1250) — matrix operand order repaired and checked by reading, not compiled, not run on hardware
//
#include <hip/hip_runtime.h>
#include <hip/hip_bf16.h>

// ---------------------------------------------------------------------------
// MeshGraphNet on MI455X (gfx1250): bf16 WMMA GEMMs, fused MLP+LN kernels,
// f32 activations in HBM, atomic scatter-add for segment_sum.
// ---------------------------------------------------------------------------

typedef __attribute__((ext_vector_type(16))) __bf16      v16bf;
typedef __attribute__((ext_vector_type(8)))  float        v8f;
typedef __attribute__((ext_vector_type(4)))  unsigned int u32x4;
typedef __attribute__((ext_vector_type(4)))  float        f32x4;

union BFrag { u32x4 u[2]; v16bf v; };

__device__ __forceinline__ unsigned short f2bf(float f) {
  unsigned int u = __float_as_uint(f);
  u += 0x7FFFu + ((u >> 16) & 1u);          // round-to-nearest-even
  return (unsigned short)(u >> 16);
}
__device__ __forceinline__ float bf2f(unsigned short u) {
  return __uint_as_float(((unsigned int)u) << 16);
}

// One WMMA step over a 32-wide K slice: A tile rows rt*16..+16 from LDS
// (row-major bf16, stride aStride), B from global pre-transposed weights
// Wt[dout][Kpad] bf16. acc[4] covers n = ch*64 .. ch*64+63.
__device__ __forceinline__ void wmma_k32(
    const unsigned short* aRow,   // &aLds[(rt*16+n16)*aStride + k0]
    const unsigned short* Wt,     // weight base
    long wRowOff,                 // k0 offset inside a weight row
    int Kpad, int rt, int ch, int kh, int n16, v8f acc[4]) {
  // A 16x32 bf16 layout: lanes 0-15 -> M=0-15; VGPR0-3: K = kh*8 + 0..7,
  // VGPR4-7: K = 16 + kh*8 + 0..7  (two b128 LDS loads)
  BFrag A;
  A.u[0] = *(const u32x4*)(aRow + kh * 8);
  A.u[1] = *(const u32x4*)(aRow + 16 + kh * 8);
#pragma unroll
  for (int t = 0; t < 4; ++t) {
    int n = ch * 64 + t * 16 + n16;
    // B 32x16 bf16 layout: lane holds column n; VGPR0-3: K = kh*8 + 0..7,
    // VGPR4-7: K = 16 + kh*8 + 0..7 (matches A).
    const unsigned short* bp = Wt + (long)n * Kpad + wRowOff + kh * 8;
    BFrag B;
    B.u[0] = *(const u32x4*)(bp);
    B.u[1] = *(const u32x4*)(bp + 16);
    acc[t] = __builtin_amdgcn_wmma_f32_16x16x32_bf16(
        false, A.v, false, B.v, (short)0, acc[t], false, false);
  }
}

__device__ __forceinline__ void acc_zero(v8f acc[4]) {
  v8f z = {0.f, 0.f, 0.f, 0.f, 0.f, 0.f, 0.f, 0.f};
#pragma unroll
  for (int t = 0; t < 4; ++t) acc[t] = z;
}

// 128->128 GEMM from an LDS bf16 activation tile (stride 128).
__device__ __forceinline__ void gemm128(
    const unsigned short* aLds, const unsigned short* Wt,
    int rt, int ch, int kh, int n16, v8f acc[4]) {
#pragma unroll
  for (int kt = 0; kt < 4; ++kt)
    wmma_k32(aLds + (rt * 16 + n16) * 128 + (kt << 5), Wt, (kt << 5),
             128, rt, ch, kh, n16, acc);
}

// Epilogue: bias (+ReLU), f32->bf16, scatter into LDS tile [64][128].
// f32 C/D layout: lane n = t*16 + (lane&15) (+ch*64); VGPR v -> m = v + kh*8.
__device__ __forceinline__ void epi_bf16(
    v8f acc[4], const float* bias, unsigned short* dst,
    int rt, int ch, int kh, int n16) {
#pragma unroll
  for (int t = 0; t < 4; ++t) {
    int n = ch * 64 + t * 16 + n16;
    float bv = bias[n];
#pragma unroll
    for (int v = 0; v < 8; ++v) {
      int m = rt * 16 + kh * 8 + v;
      dst[m * 128 + n] = f2bf(fmaxf(acc[t][v] + bv, 0.f));
    }
  }
}

// MODE 0: encoder  (in = normalize(raw), K1=32 padded from din)
// MODE 1: edge layer (in = [h[dst] | h[src] | e], K1=384; out: e += LN(mlp))
// MODE 2: node layer (in = [h | agg], K1=256; out: h = LN(mlp) + h)
// MODE 3: decoder   (in = h, K1=128; relu,relu, then 128->2 dot; no LN)
template <int MODE>
__global__ __launch_bounds__(256) void fused_mlp(
    int nrows, int K1,
    const float* raw, const float* meanv, const float* stdv, int din,
    const float* hbuf, const float* ebuf, const float* aggbuf,
    const int* srcIdx, const int* dstIdx,
    const unsigned short* W1t, const float* b1,
    const unsigned short* W2t, const float* b2,
    const unsigned short* W3t, const float* b3,
    const float* lng, const float* lnb,
    const float* w3raw, const float* b3raw,
    float* out) {
  __shared__ __align__(16) unsigned short smem[3 * 64 * 128];  // 48 KB
  unsigned short* achunk = smem;                // [64][<=128] gemm1 input chunk
  unsigned short* act1 = smem + 64 * 128;       // gemm1 out / gemm2 in
  unsigned short* act2 = smem + 2 * 64 * 128;   // gemm2 out / gemm3 in
  float* fbuf = (float*)smem;                   // [64][128] f32 (aliases achunk+act1)

  const int tid = threadIdx.x;
  const int lane = tid & 31;
  const int wv = tid >> 5;          // 8 wave32s
  const int rt = wv & 3;            // row tile 0..3  (rows rt*16..+16)
  const int ch = wv >> 2;           // col half 0..1  (cols ch*64..+64)
  const int kh = lane >> 4;         // K-half within fragment layouts
  const int n16 = lane & 15;
  const long base = (long)blockIdx.x * 64;

  const int chunkK = (K1 < 128) ? K1 : 128;
  const int nchunks = K1 / chunkK;

  v8f acc[4];
  acc_zero(acc);

  // ---- layer 1: stream input in K-chunks of <=128 through LDS ----
  for (int kc = 0; kc < nchunks; ++kc) {
    const int c4n = chunkK >> 2;
    for (int i = tid; i < 64 * c4n; i += 256) {
      int r = i / c4n;
      int c = (i - r * c4n) << 2;
      long grow = base + r;
      f32x4 vv = {0.f, 0.f, 0.f, 0.f};
      if (grow < nrows) {
        if (MODE == 0) {
#pragma unroll
          for (int q = 0; q < 4; ++q) {
            int kk = c + q;
            if (kk < din) vv[q] = (raw[grow * din + kk] - meanv[kk]) / stdv[kk];
          }
        } else if (MODE == 1) {
          const float* sp;
          if (kc == 0)      sp = hbuf + (long)dstIdx[grow] * 128;
          else if (kc == 1) sp = hbuf + (long)srcIdx[grow] * 128;
          else              sp = ebuf + grow * 128;
          vv = *(const f32x4*)(sp + c);
        } else if (MODE == 2) {
          const float* sp = (kc == 0 ? hbuf : aggbuf) + grow * 128;
          vv = *(const f32x4*)(sp + c);
        } else {
          vv = *(const f32x4*)(hbuf + grow * 128 + c);
        }
      }
      unsigned short* dp = achunk + r * chunkK + c;
      dp[0] = f2bf(vv.x); dp[1] = f2bf(vv.y);
      dp[2] = f2bf(vv.z); dp[3] = f2bf(vv.w);
    }
    __syncthreads();
    const int ktiles = chunkK >> 5;
    for (int kt = 0; kt < ktiles; ++kt)
      wmma_k32(achunk + (rt * 16 + n16) * chunkK + (kt << 5), W1t,
               (long)kc * 128 + (kt << 5), K1, rt, ch, kh, n16, acc);
    __syncthreads();
  }
  epi_bf16(acc, b1, act1, rt, ch, kh, n16);
  __syncthreads();

  // ---- layer 2 ----
  acc_zero(acc);
  gemm128(act1, W2t, rt, ch, kh, n16, acc);
  epi_bf16(acc, b2, act2, rt, ch, kh, n16);
  __syncthreads();

  if (MODE != 3) {
    // ---- layer 3 (no relu) + LayerNorm ----
    acc_zero(acc);
    gemm128(act2, W3t, rt, ch, kh, n16, acc);
#pragma unroll
    for (int t = 0; t < 4; ++t) {
      int n = ch * 64 + t * 16 + n16;
      float bv = b3[n];
#pragma unroll
      for (int v = 0; v < 8; ++v) {
        int m = rt * 16 + kh * 8 + v;
        fbuf[m * 128 + n] = acc[t][v] + bv;   // fbuf aliases dead achunk/act1
      }
    }
    __syncthreads();
    if (tid < 64) {
      long grow = base + tid;
      if (grow < nrows) {
        const float* row = fbuf + tid * 128;
        float s = 0.f, s2 = 0.f;
#pragma unroll 4
        for (int c = 0; c < 128; ++c) { float x = row[c]; s += x; s2 += x * x; }
        float mu = s * (1.f / 128.f);
        float var = s2 * (1.f / 128.f) - mu * mu;
        float rs = rsqrtf(var + 1e-5f);
#pragma unroll 4
        for (int c = 0; c < 128; ++c) {
          float val = (row[c] - mu) * rs * lng[c] + lnb[c];
          if (MODE == 0)      out[grow * 128 + c] = val;
          else if (MODE == 1) out[grow * 128 + c] = ebuf[grow * 128 + c] + val;
          else                out[grow * 128 + c] = hbuf[grow * 128 + c] + val;
        }
      }
    }
  } else {
    // ---- decoder head: 128 -> 2 plain dot from LDS ----
    if (tid < 128) {
      int r = tid >> 1, c = tid & 1;
      long grow = base + r;
      if (grow < nrows) {
        float s = b3raw[c];
        const unsigned short* ar = act2 + r * 128;
#pragma unroll 4
        for (int k = 0; k < 128; ++k) s += bf2f(ar[k]) * w3raw[k * 2 + c];
        out[grow * 2 + c] = s;
      }
    }
  }
}

// Transpose + pad weights (f32 [din][dout]) -> bf16 [dout][Kpad] for B frags.
__global__ __launch_bounds__(256) void prep_w(const float* W,
                                              unsigned short* Wt, int din,
                                              int dout, int Kpad) {
  int i = blockIdx.x * 256 + threadIdx.x;
  if (i >= dout * Kpad) return;
  int n = i / Kpad, k = i - n * Kpad;
  float v = (k < din) ? W[(long)k * dout + n] : 0.f;
  Wt[i] = f2bf(v);
}

__global__ __launch_bounds__(256) void zero_f32(float* p, long n) {
  long i = (long)blockIdx.x * 256 + threadIdx.x;
  long stride = (long)gridDim.x * 256;
  for (; i < n; i += stride) p[i] = 0.f;
}

// segment_sum(e, src) via f32 atomics; one edge handled by 32 lanes (4 cols ea).
__global__ __launch_bounds__(256) void scatter_add_kernel(const float* e,
                                                          const int* src,
                                                          float* agg, int ne) {
  long t = (long)blockIdx.x * 256 + threadIdx.x;
  long j = t >> 5;
  if (j >= ne) return;
  int c = (int)(t & 31) << 2;
  if (j + 8 < ne) __builtin_prefetch(e + (j + 8) * 128 + c, 0, 1);
  int s = src[j];
  f32x4 v = *(const f32x4*)(e + j * 128 + c);
  float* a = agg + (long)s * 128 + c;
  unsafeAtomicAdd(a + 0, v.x);
  unsafeAtomicAdd(a + 1, v.y);
  unsafeAtomicAdd(a + 2, v.z);
  unsafeAtomicAdd(a + 3, v.w);
}

extern "C" void kernel_launch(void* const* d_in, const int* in_sizes, int n_in,
                              void* d_out, int out_size, void* d_ws,
                              size_t ws_size, hipStream_t stream) {
  const int Nn = 100000, Ee = 600000, Lyr = 6;

  // workspace carve-up
  char* ws = (char*)d_ws;
  float* h   = (float*)(ws);                       //  51.2 MB
  float* e   = (float*)(ws + 51200000LL);          // 307.2 MB
  float* agg = (float*)(ws + 358400000LL);         //  51.2 MB
  unsigned short* wbase = (unsigned short*)(ws + 409600000LL);  // ~2 MB bf16 W

  const float* x      = (const float*)d_in[0];
  const int*   eidx   = (const int*)d_in[1];
  const float* eattr  = (const float*)d_in[2];
  const float* mean_x = (const float*)d_in[3];
  const float* std_x  = (const float*)d_in[4];
  const float* mean_e = (const float*)d_in[5];
  const float* std_e  = (const float*)d_in[6];
  const int* srcI = eidx;        // edge_index[0]
  const int* dstI = eidx + Ee;   // edge_index[1]

  auto P = [&](int i) { return (const float*)d_in[i]; };

  // JAX pytree (sorted keys) leaf order:
  // 7..12  decoder   {l1.b,l1.w,l2.b,l2.w,l3.b,l3.w}
  // 13..20 edge_enc  {l1.b,l1.w,l2.b,l2.w,l3.b,l3.w,ln_b,ln_g}
  // 21..68 edge_layers[0..5] (8 each)
  // 69..76 node_enc
  // 77..124 node_layers[0..5]
  size_t wo = 0;
  auto prep = [&](int widx, int din, int Kpad) -> unsigned short* {
    unsigned short* wt = wbase + wo;
    wo += (size_t)Kpad * 128;
    int total = 128 * Kpad;
    prep_w<<<(total + 255) / 256, 256, 0, stream>>>(P(widx), wt, din, 128, Kpad);
    return wt;
  };

  unsigned short* ne_w1 = prep(70, 11, 32);
  unsigned short* ne_w2 = prep(72, 128, 128);
  unsigned short* ne_w3 = prep(74, 128, 128);
  unsigned short* ee_w1 = prep(14, 3, 32);
  unsigned short* ee_w2 = prep(16, 128, 128);
  unsigned short* ee_w3 = prep(18, 128, 128);
  unsigned short *el_w1[6], *el_w2[6], *el_w3[6], *nl_w1[6], *nl_w2[6], *nl_w3[6];
  for (int i = 0; i < Lyr; ++i) {
    int b = 21 + 8 * i;
    el_w1[i] = prep(b + 1, 384, 384);
    el_w2[i] = prep(b + 3, 128, 128);
    el_w3[i] = prep(b + 5, 128, 128);
  }
  for (int i = 0; i < Lyr; ++i) {
    int b = 77 + 8 * i;
    nl_w1[i] = prep(b + 1, 256, 256);
    nl_w2[i] = prep(b + 3, 128, 128);
    nl_w3[i] = prep(b + 5, 128, 128);
  }
  unsigned short* dc_w1 = prep(8, 128, 128);
  unsigned short* dc_w2 = prep(10, 128, 128);

  dim3 thr(256);
  dim3 blkN((Nn + 63) / 64), blkE((Ee + 63) / 64);

  // encoders
  fused_mlp<0><<<blkN, thr, 0, stream>>>(
      Nn, 32, x, mean_x, std_x, 11, nullptr, nullptr, nullptr, nullptr, nullptr,
      ne_w1, P(69), ne_w2, P(71), ne_w3, P(73), P(76), P(75),
      nullptr, nullptr, h);
  fused_mlp<0><<<blkE, thr, 0, stream>>>(
      Ee, 32, eattr, mean_e, std_e, 3, nullptr, nullptr, nullptr, nullptr, nullptr,
      ee_w1, P(13), ee_w2, P(15), ee_w3, P(17), P(20), P(19),
      nullptr, nullptr, e);

  // message-passing layers
  for (int i = 0; i < Lyr; ++i) {
    int b = 21 + 8 * i;
    fused_mlp<1><<<blkE, thr, 0, stream>>>(
        Ee, 384, nullptr, nullptr, nullptr, 0, h, e, nullptr, srcI, dstI,
        el_w1[i], P(b + 0), el_w2[i], P(b + 2), el_w3[i], P(b + 4),
        P(b + 7), P(b + 6), nullptr, nullptr, e);

    zero_f32<<<2048, thr, 0, stream>>>(agg, (long)Nn * 128);
    long st = (long)Ee * 32;
    scatter_add_kernel<<<(unsigned)((st + 255) / 256), thr, 0, stream>>>(
        e, srcI, agg, Ee);

    int nb = 77 + 8 * i;
    fused_mlp<2><<<blkN, thr, 0, stream>>>(
        Nn, 256, nullptr, nullptr, nullptr, 0, h, nullptr, agg, nullptr, nullptr,
        nl_w1[i], P(nb + 0), nl_w2[i], P(nb + 2), nl_w3[i], P(nb + 4),
        P(nb + 7), P(nb + 6), nullptr, nullptr, h);
  }

  // decoder
  fused_mlp<3><<<blkN, thr, 0, stream>>>(
      Nn, 128, nullptr, nullptr, nullptr, 0, h, nullptr, nullptr, nullptr,
      nullptr, dc_w1, P(7), dc_w2, P(9), nullptr, nullptr, nullptr, nullptr,
      P(12), P(11), (float*)d_out);
}
